// MultiScaleSE3Simple_79748952752293
// MI455X (gfx1250) — compile-verified, matrix operand order checked by reading, not run on hardware
//
#include <hip/hip_runtime.h>
#include <stdint.h>

// ---------------------------------------------------------------------------
// Problem constants (from reference)
// ---------------------------------------------------------------------------
#define NN      50000
#define EE      800000
#define D_IN    64
#define D_H     128
#define LN_EPS  1e-5f
#define WAVES   4           // waves per block (blockDim = 128 = 4 x wave32)
#define NTILES  8           // 128 / 16 output column tiles
#define MTILES  3125        // 50000 / 16 node row tiles
#define ETILES  50000       // 800000 / 16 edge row tiles

// ---------------------------------------------------------------------------
// Vector types for WMMA (CDNA5 wave32)
// ---------------------------------------------------------------------------
typedef __attribute__((ext_vector_type(16))) __bf16 v16bf;
typedef __attribute__((ext_vector_type(8)))  __bf16 v8bf;
typedef __attribute__((ext_vector_type(8)))  float  v8f;

// ---------------------------------------------------------------------------
// bf16 helpers — use native __bf16 conversions so clang emits the gfx1250
// hardware cvt ops (packed where possible) instead of bfe/add3/mov_b16 chains.
// ---------------------------------------------------------------------------
__device__ __forceinline__ __bf16 f2bf(float f) { return (__bf16)f; }

__device__ __forceinline__ unsigned pack2bf(float a, float b) {
  __bf16 x = (__bf16)a, y = (__bf16)b;
  unsigned short ux = __builtin_bit_cast(unsigned short, x);
  unsigned short uy = __builtin_bit_cast(unsigned short, y);
  return (unsigned)ux | ((unsigned)uy << 16);
}

__device__ __forceinline__ __bf16 bf_zero() {
  unsigned short z = 0;
  return __builtin_bit_cast(__bf16, z);
}

__device__ __forceinline__ v16bf combine8(v8bf lo, v8bf hi) {
  v16bf r;
#pragma unroll
  for (int i = 0; i < 8; i++) { r[i] = lo[i]; r[8 + i] = hi[i]; }
  return r;
}

// A-fragment for 16x16x32 bf16: lane (m = lane&15, half = lane>>4) holds
// K = kt*32 + half*8 + {0..7}  and  K = kt*32 + 16 + half*8 + {0..7}
__device__ __forceinline__ v16bf load_row_frag(const __bf16* row, int kt, int half) {
  const __bf16* p = row + kt * 32 + half * 8;
  v8bf lo = *(const v8bf*)p;
  v8bf hi = *(const v8bf*)(p + 16);
  return combine8(lo, hi);
}

// A-fragment built from an fp32 row (native packed converts)
__device__ __forceinline__ v16bf cvt_row_frag(const float* row, int kt, int half) {
  const float* p = row + kt * 32 + half * 8;
  v16bf a;
#pragma unroll
  for (int j = 0; j < 8; j++) { a[j] = f2bf(p[j]); a[8 + j] = f2bf(p[16 + j]); }
  return a;
}

__device__ __forceinline__ v8f wmma_bf16(v16bf a, v16bf b, v8f c) {
  // (neg_a, A, neg_b, B, c_mod, C, reuse_a, reuse_b) -> v_wmma_f32_16x16x32_bf16
  return __builtin_amdgcn_wmma_f32_16x16x32_bf16(false, a, false, b, (short)0, c,
                                                 false, false);
}

__device__ __forceinline__ void zero_acc(v8f acc[NTILES]) {
#pragma unroll
  for (int nt = 0; nt < NTILES; nt++)
#pragma unroll
    for (int i = 0; i < 8; i++) acc[nt][i] = 0.f;
}

// One A-fragment against all 8 packed-weight B fragments of K-tile `Wk`.
__device__ __forceinline__ void mma_row(v8f acc[NTILES], v16bf a,
                                        const __bf16* Wk, int lane) {
#pragma unroll
  for (int nt = 0; nt < NTILES; nt++) {
    v16bf b = *(const v16bf*)(Wk + (size_t)nt * 512 + (size_t)lane * 16);
    acc[nt] = wmma_bf16(a, b, acc[nt]);
  }
}

// C/D layout: lane<16 -> N=lane, M = vgpr r; lane>=16 -> N=lane-16, M = 8+r.
__device__ __forceinline__ void store_frags_bias(float* st, const v8f acc[NTILES],
                                                 const float* bias, int lane) {
  int col0 = lane & 15, base = (lane >> 4) * 8;
#pragma unroll
  for (int nt = 0; nt < NTILES; nt++) {
    float bv = bias[nt * 16 + col0];
#pragma unroll
    for (int r = 0; r < 8; r++)
      st[(base + r) * 132 + nt * 16 + col0] = acc[nt][r] + bv;
  }
}

// Single-pass LN stats (sum + sum of squares, float4-vectorized)
__device__ __forceinline__ void row_stats(const float* st, float* mu, float* rs,
                                          int lane) {
  if (lane < 16) {
    const float* row = st + lane * 132;
    float s = 0.f, ss = 0.f;
#pragma unroll
    for (int k = 0; k < 128; k += 4) {
      float4 q = *(const float4*)(row + k);
      s  += q.x + q.y + q.z + q.w;
      ss += q.x * q.x + q.y * q.y + q.z * q.z + q.w * q.w;
    }
    float m = s * (1.f / 128.f);
    float v = ss * (1.f / 128.f) - m * m;
    mu[lane] = m;
    rs[lane] = rsqrtf(fmaxf(v, 0.f) + LN_EPS);
  }
}

// ---------------------------------------------------------------------------
// Utility kernels
// ---------------------------------------------------------------------------
__global__ void zero_f32_kernel(float* p, int n) {
  int i = blockIdx.x * blockDim.x + threadIdx.x;
  if (i < n) p[i] = 0.f;
}

// Pack fp32 weight W[K x 128] into per-lane WMMA B-fragments (bf16), zero-pad K.
__global__ void pack_w_kernel(const float* __restrict__ w, __bf16* __restrict__ out,
                              int K, int Kt) {
  int gid = blockIdx.x * blockDim.x + threadIdx.x;
  int total = Kt * NTILES * 32;
  if (gid >= total) return;
  int lane = gid & 31, f = gid >> 5;
  int kt = f >> 3, nt = f & 7;
  int half = lane >> 4, n = nt * 16 + (lane & 15);
  __bf16* o = out + (size_t)f * 512 + (size_t)lane * 16;
#pragma unroll
  for (int j = 0; j < 8; j++) {
    int k0 = kt * 32 + half * 8 + j;
    int k1 = k0 + 16;
    float v0 = (k0 < K) ? w[(size_t)k0 * 128 + n] : 0.f;
    float v1 = (k1 < K) ? w[(size_t)k1 * 128 + n] : 0.f;
    o[j]     = f2bf(v0);
    o[8 + j] = f2bf(v1);
  }
}

// ---------------------------------------------------------------------------
// Embedding: h = relu(LN(x @ We + be)) ; writes h (f32) and hb (bf16)
// ---------------------------------------------------------------------------
__global__ void __launch_bounds__(128)
embed_kernel(const float* __restrict__ x, const __bf16* __restrict__ Wf,
             const float* __restrict__ bias, const float* __restrict__ g,
             const float* __restrict__ lb, float* __restrict__ h,
             __bf16* __restrict__ hb, int Mtiles) {
  __shared__ float s_tile[WAVES][16 * 132];
  __shared__ float s_mu[WAVES][16], s_rs[WAVES][16];
  int wave = threadIdx.x >> 5, lane = threadIdx.x & 31;
  int tile = blockIdx.x * WAVES + wave;
  bool active = tile < Mtiles;
  int m = lane & 15, half = lane >> 4;
  v8f acc[NTILES];
  zero_acc(acc);
  if (active) {
    const float* xr = x + (size_t)(tile * 16 + m) * D_IN;
#pragma unroll
    for (int kt = 0; kt < 2; kt++)  // K = 64 -> 2 K-tiles
      mma_row(acc, cvt_row_frag(xr, kt, half), Wf + (size_t)kt * 4096, lane);
    store_frags_bias(&s_tile[wave][0], acc, bias, lane);
  }
  __syncthreads();
  if (active) row_stats(&s_tile[wave][0], s_mu[wave], s_rs[wave], lane);
  __syncthreads();
  if (active) {
    for (int idx = lane * 2; idx < 2048; idx += 64) {
      int row = idx >> 7, col = idx & 127;
      float sc = s_rs[wave][row], mn = s_mu[wave][row];
      float v0 = fmaxf((s_tile[wave][row * 132 + col]     - mn) * sc * g[col]     + lb[col],     0.f);
      float v1 = fmaxf((s_tile[wave][row * 132 + col + 1] - mn) * sc * g[col + 1] + lb[col + 1], 0.f);
      size_t o = (size_t)(tile * 16 + row) * 128 + col;
      float2 t; t.x = v0; t.y = v1;
      *(float2*)&h[o] = t;
      *(unsigned*)&hb[o] = pack2bf(v0, v1);
    }
  }
}

// ---------------------------------------------------------------------------
// Node GEMM: MODE 0 = +bias, LN, ReLU -> bf16 ; MODE 1 = +bias -> bf16
// ---------------------------------------------------------------------------
template <int MODE>
__global__ void __launch_bounds__(128)
gemm_node_kernel(const __bf16* __restrict__ A, const __bf16* __restrict__ Wf,
                 const float* __restrict__ bias, const float* __restrict__ g,
                 const float* __restrict__ lb, __bf16* __restrict__ outb,
                 int Mtiles) {
  __shared__ float s_tile[WAVES][16 * 132];
  __shared__ float s_mu[WAVES][16], s_rs[WAVES][16];
  int wave = threadIdx.x >> 5, lane = threadIdx.x & 31;
  int tile = blockIdx.x * WAVES + wave;
  bool active = tile < Mtiles;
  int m = lane & 15, half = lane >> 4;
  v8f acc[NTILES];
  zero_acc(acc);
  if (active) {
    const __bf16* Ar = A + (size_t)(tile * 16 + m) * 128;
#pragma unroll
    for (int kt = 0; kt < 4; kt++)
      mma_row(acc, load_row_frag(Ar, kt, half), Wf + (size_t)kt * 4096, lane);
  }
  if constexpr (MODE == 1) {
    if (!active) return;
    int col0 = lane & 15, base = half * 8;
#pragma unroll
    for (int nt = 0; nt < NTILES; nt++) {
      int col = nt * 16 + col0;
      float bv = bias[col];
#pragma unroll
      for (int r = 0; r < 8; r++)
        outb[(size_t)(tile * 16 + base + r) * 128 + col] = f2bf(acc[nt][r] + bv);
    }
    return;
  } else {
    if (active) store_frags_bias(&s_tile[wave][0], acc, bias, lane);
    __syncthreads();
    if (active) row_stats(&s_tile[wave][0], s_mu[wave], s_rs[wave], lane);
    __syncthreads();
    if (active) {
      for (int idx = lane * 2; idx < 2048; idx += 64) {
        int row = idx >> 7, col = idx & 127;
        float sc = s_rs[wave][row], mn = s_mu[wave][row];
        float v0 = fmaxf((s_tile[wave][row * 132 + col]     - mn) * sc * g[col]     + lb[col],     0.f);
        float v1 = fmaxf((s_tile[wave][row * 132 + col + 1] - mn) * sc * g[col + 1] + lb[col + 1], 0.f);
        *(unsigned*)&outb[(size_t)(tile * 16 + row) * 128 + col] = pack2bf(v0, v1);
      }
    }
  }
}

// ---------------------------------------------------------------------------
// Fused edge MLP: gather h[src],h[dst],edge_attr -> lin1 -> LN -> relu
//                 -> lin2 -> atomic scatter-add into h_agg[dst]
// K1 padded 261 -> 288 (9 K-tiles: 4 src, 4 dst, 1 attr). ETILES % WAVES == 0.
// ---------------------------------------------------------------------------
__global__ void __launch_bounds__(128)
edge_kernel(const __bf16* __restrict__ hb, const int* __restrict__ src,
            const int* __restrict__ dst, const float* __restrict__ eattr,
            const __bf16* __restrict__ W1, const float* __restrict__ b1,
            const float* __restrict__ g1, const float* __restrict__ lb1,
            const __bf16* __restrict__ W2, const float* __restrict__ b2,
            float* __restrict__ hagg) {
  __shared__ float  s_tile[WAVES][16 * 132];
  __shared__ __bf16 s_mid[WAVES][16 * 136];
  __shared__ float  s_mu[WAVES][16], s_rs[WAVES][16];
  __shared__ int    s_src[WAVES][16], s_dst[WAVES][16];
  int wave = threadIdx.x >> 5, lane = threadIdx.x & 31;
  int tile = blockIdx.x * WAVES + wave;
  int m = lane & 15, half = lane >> 4;
  if (lane < 16) {
    int e = tile * 16 + lane;
    s_src[wave][lane] = src[e];
    s_dst[wave][lane] = dst[e];
    __builtin_prefetch(eattr + (size_t)e * 5, 0, 0);
  }
  __syncthreads();
  int sN = s_src[wave][m], dN = s_dst[wave][m];
  int erow = tile * 16 + m;

  v8f acc[NTILES];
  zero_acc(acc);
  const __bf16* rsrc = hb + (size_t)sN * 128;
  const __bf16* rdst = hb + (size_t)dN * 128;
#pragma unroll
  for (int kt = 0; kt < 4; kt++)
    mma_row(acc, load_row_frag(rsrc, kt, half), W1 + (size_t)kt * 4096, lane);
#pragma unroll
  for (int kt = 0; kt < 4; kt++)
    mma_row(acc, load_row_frag(rdst, kt, half), W1 + (size_t)(4 + kt) * 4096, lane);
  {  // edge_attr tile: K = 256..260 live, rest zero
    v16bf a;
    __bf16 z = bf_zero();
#pragma unroll
    for (int i = 0; i < 16; i++) a[i] = z;
    if (half == 0) {
      const float* ap = eattr + (size_t)erow * 5;
#pragma unroll
      for (int i = 0; i < 5; i++) a[i] = f2bf(ap[i]);
    }
    mma_row(acc, a, W1 + (size_t)8 * 4096, lane);
  }

  store_frags_bias(&s_tile[wave][0], acc, b1, lane);
  __syncthreads();
  row_stats(&s_tile[wave][0], s_mu[wave], s_rs[wave], lane);
  __syncthreads();
  for (int idx = lane * 2; idx < 2048; idx += 64) {
    int row = idx >> 7, col = idx & 127;
    float sc = s_rs[wave][row], mn = s_mu[wave][row];
    float v0 = fmaxf((s_tile[wave][row * 132 + col]     - mn) * sc * g1[col]     + lb1[col],     0.f);
    float v1 = fmaxf((s_tile[wave][row * 132 + col + 1] - mn) * sc * g1[col + 1] + lb1[col + 1], 0.f);
    *(unsigned*)&s_mid[wave][row * 136 + col] = pack2bf(v0, v1);
  }
  __syncthreads();

  // lin2: A from LDS (stride 136 keeps 16B alignment per K-group)
  v8f acc2[NTILES];
  zero_acc(acc2);
  const __bf16* Ar = &s_mid[wave][m * 136];
#pragma unroll
  for (int kt = 0; kt < 4; kt++) {
    const __bf16* p = Ar + kt * 32 + half * 8;
    v8bf lo = *(const v8bf*)p;
    v8bf hi = *(const v8bf*)(p + 16);
    mma_row(acc2, combine8(lo, hi), W2 + (size_t)kt * 4096, lane);
  }

  int col0 = m, base = half * 8;
#pragma unroll
  for (int nt = 0; nt < NTILES; nt++) {
    int col = nt * 16 + col0;
    float bv = b2[col];
#pragma unroll
    for (int r = 0; r < 8; r++)
      atomicAdd(&hagg[(size_t)s_dst[wave][base + r] * 128 + col],
                acc2[nt][r] + bv);
  }
}

// ---------------------------------------------------------------------------
// Update: h = LN( [hn, h_agg] @ Wu + bu + h_identity ); writes h f32 + bf16
// ---------------------------------------------------------------------------
__global__ void __launch_bounds__(128)
update_kernel(const __bf16* __restrict__ hnb, const float* __restrict__ hagg,
              const __bf16* __restrict__ Wu, const float* __restrict__ bias,
              const float* __restrict__ g, const float* __restrict__ lb,
              float* __restrict__ h, __bf16* __restrict__ hb, int Mtiles) {
  __shared__ float s_tile[WAVES][16 * 132];
  __shared__ float s_mu[WAVES][16], s_rs[WAVES][16];
  int wave = threadIdx.x >> 5, lane = threadIdx.x & 31;
  int tile = blockIdx.x * WAVES + wave;
  bool active = tile < Mtiles;
  int m = lane & 15, half = lane >> 4;
  v8f acc[NTILES];
  zero_acc(acc);
  if (active) {
    const __bf16* Ar = hnb + (size_t)(tile * 16 + m) * 128;
#pragma unroll
    for (int kt = 0; kt < 4; kt++)
      mma_row(acc, load_row_frag(Ar, kt, half), Wu + (size_t)kt * 4096, lane);
    const float* Gr = hagg + (size_t)(tile * 16 + m) * 128;
#pragma unroll
    for (int kt = 0; kt < 4; kt++)
      mma_row(acc, cvt_row_frag(Gr, kt, half), Wu + (size_t)(4 + kt) * 4096, lane);
    // bias + residual into LDS
    int col0 = m, base = half * 8;
#pragma unroll
    for (int nt = 0; nt < NTILES; nt++) {
      int col = nt * 16 + col0;
      float bv = bias[col];
#pragma unroll
      for (int r = 0; r < 8; r++)
        s_tile[wave][(base + r) * 132 + col] =
            acc[nt][r] + bv + h[(size_t)(tile * 16 + base + r) * 128 + col];
    }
  }
  __syncthreads();
  if (active) row_stats(&s_tile[wave][0], s_mu[wave], s_rs[wave], lane);
  __syncthreads();
  if (active) {
    for (int idx = lane * 2; idx < 2048; idx += 64) {
      int row = idx >> 7, col = idx & 127;
      float sc = s_rs[wave][row], mn = s_mu[wave][row];
      float v0 = (s_tile[wave][row * 132 + col]     - mn) * sc * g[col]     + lb[col];
      float v1 = (s_tile[wave][row * 132 + col + 1] - mn) * sc * g[col + 1] + lb[col + 1];
      size_t o = (size_t)(tile * 16 + row) * 128 + col;
      float2 t; t.x = v0; t.y = v1;
      *(float2*)&h[o] = t;
      *(unsigned*)&hb[o] = pack2bf(v0, v1);
    }
  }
}

// ---------------------------------------------------------------------------
// Output head: node_emb = LN(h @ Wo + bo) -> d_out, plus graph pooling atomics
// ---------------------------------------------------------------------------
__global__ void __launch_bounds__(128)
out_kernel(const __bf16* __restrict__ hb, const __bf16* __restrict__ Wo,
           const float* __restrict__ bias, const float* __restrict__ g,
           const float* __restrict__ lb, const int* __restrict__ batch,
           float* __restrict__ node_emb, float* __restrict__ pool_add,
           float* __restrict__ pool_cnt, int Mtiles) {
  __shared__ float s_tile[WAVES][16 * 132];
  __shared__ float s_mu[WAVES][16], s_rs[WAVES][16];
  __shared__ int   s_b[WAVES][16];
  int wave = threadIdx.x >> 5, lane = threadIdx.x & 31;
  int tile = blockIdx.x * WAVES + wave;
  bool active = tile < Mtiles;
  int m = lane & 15, half = lane >> 4;
  v8f acc[NTILES];
  zero_acc(acc);
  if (active) {
    if (lane < 16) s_b[wave][lane] = batch[tile * 16 + lane];
    const __bf16* Ar = hb + (size_t)(tile * 16 + m) * 128;
#pragma unroll
    for (int kt = 0; kt < 4; kt++)
      mma_row(acc, load_row_frag(Ar, kt, half), Wo + (size_t)kt * 4096, lane);
    store_frags_bias(&s_tile[wave][0], acc, bias, lane);
  }
  __syncthreads();
  if (active) row_stats(&s_tile[wave][0], s_mu[wave], s_rs[wave], lane);
  __syncthreads();
  if (active) {
    for (int idx = lane * 2; idx < 2048; idx += 64) {
      int row = idx >> 7, col = idx & 127;
      float sc = s_rs[wave][row], mn = s_mu[wave][row];
      float v0 = (s_tile[wave][row * 132 + col]     - mn) * sc * g[col]     + lb[col];
      float v1 = (s_tile[wave][row * 132 + col + 1] - mn) * sc * g[col + 1] + lb[col + 1];
      size_t o = (size_t)(tile * 16 + row) * 128 + col;
      float2 t; t.x = v0; t.y = v1;
      *(float2*)&node_emb[o] = t;
      int b = s_b[wave][row];
      atomicAdd(&pool_add[(size_t)b * 128 + col],     v0);
      atomicAdd(&pool_add[(size_t)b * 128 + col + 1], v1);
    }
    if (lane < 16) atomicAdd(&pool_cnt[s_b[wave][lane]], 1.0f);
  }
}

// ---------------------------------------------------------------------------
// Final small GEMM: graph_emb = [mean_pool, add_pool] @ pool.w + pool.b
// ---------------------------------------------------------------------------
__global__ void pool_kernel(const float* __restrict__ pool_add,
                            const float* __restrict__ pool_cnt,
                            const float* __restrict__ pw,
                            const float* __restrict__ pb,
                            float* __restrict__ graph_emb) {
  int col = threadIdx.x;
  if (col >= 128) return;
  for (int gi = 0; gi < 16; gi++) {
    float inv = 1.f / fmaxf(pool_cnt[gi], 1.f);
    float sum = pb[col];
    for (int k = 0; k < 128; k++) {
      float av = pool_add[gi * 128 + k];
      sum += (av * inv) * pw[(size_t)k * 128 + col];       // mean part
      sum += av * pw[(size_t)(128 + k) * 128 + col];       // add  part
    }
    graph_emb[gi * 128 + col] = sum;
  }
}

// ---------------------------------------------------------------------------
// Host launcher
// ---------------------------------------------------------------------------
extern "C" void kernel_launch(void* const* d_in, const int* in_sizes, int n_in,
                              void* d_out, int out_size, void* d_ws, size_t ws_size,
                              hipStream_t stream) {
  (void)in_sizes; (void)n_in; (void)out_size; (void)ws_size;

  const float* x     = (const float*)d_in[0];
  const int*   eidx  = (const int*)d_in[1];
  const int*   srcI  = eidx;
  const int*   dstI  = eidx + EE;
  const float* eattr = (const float*)d_in[2];
  const int*   batch = (const int*)d_in[4];   // d_in[3] = pos (unused)

  auto P = [&](int i) -> const float* { return (const float*)d_in[i]; };
  // param leaf indices (setup_inputs insertion order):
  // 5..8 embed{lin.w,lin.b,ln.g,ln.b}; layer l base = 9+16*l:
  //  +0/1 node_lin1 w/b, +2/3 node_ln g/b, +4/5 node_lin2 w/b,
  //  +6/7 edge_lin1 w/b, +8/9 edge_ln g/b, +10/11 edge_lin2 w/b,
  //  +12/13 update w/b, +14/15 out_ln g/b
  // 57..60 out{lin.w,lin.b,ln.g,ln.b}; 61/62 pool w/b

  char* ws = (char*)d_ws;
  size_t off = 0;
  auto alloc = [&](size_t bytes) -> char* {
    char* p = ws + off;
    off = (off + bytes + 255) & ~(size_t)255;
    return p;
  };
  auto pack = [&](const float* w, int K) -> const __bf16* {
    int Kt = (K + 31) / 32;
    __bf16* o = (__bf16*)alloc((size_t)Kt * NTILES * 512 * sizeof(__bf16));
    int total = Kt * NTILES * 32;
    pack_w_kernel<<<(total + 255) / 256, 256, 0, stream>>>(w, o, K, Kt);
    return o;
  };

  // --- pack all weights into WMMA B-fragment layout (bf16) ---
  const __bf16* Wemb = pack(P(5), 64);
  const __bf16* Wn1[3]; const __bf16* Wn2[3]; const __bf16* We1[3];
  const __bf16* We2[3]; const __bf16* Wup[3];
  for (int l = 0; l < 3; l++) {
    int b0 = 9 + 16 * l;
    Wn1[l] = pack(P(b0 + 0), 128);
    Wn2[l] = pack(P(b0 + 4), 128);
    We1[l] = pack(P(b0 + 6), 261);   // padded to 288 (9 K-tiles)
    We2[l] = pack(P(b0 + 10), 128);
    Wup[l] = pack(P(b0 + 12), 256);
  }
  const __bf16* Wout = pack(P(57), 128);

  // --- activation buffers ---
  float*  h        = (float*)alloc((size_t)NN * 128 * 4);
  __bf16* hbuf     = (__bf16*)alloc((size_t)NN * 128 * 2);
  __bf16* midb     = (__bf16*)alloc((size_t)NN * 128 * 2);
  __bf16* hnb      = (__bf16*)alloc((size_t)NN * 128 * 2);
  float*  hagg     = (float*)alloc((size_t)NN * 128 * 4);
  float*  pool_add = (float*)alloc(16 * 128 * 4);
  float*  pool_cnt = (float*)alloc(16 * 4);

  const int nodeBlocks = (MTILES + WAVES - 1) / WAVES;  // 782
  const int edgeBlocks = ETILES / WAVES;                // 12500

  // --- embedding ---
  embed_kernel<<<nodeBlocks, 128, 0, stream>>>(x, Wemb, P(6), P(7), P(8), h, hbuf,
                                               MTILES);

  // --- message-passing layers ---
  for (int l = 0; l < 3; l++) {
    int b0 = 9 + 16 * l;
    gemm_node_kernel<0><<<nodeBlocks, 128, 0, stream>>>(
        hbuf, Wn1[l], P(b0 + 1), P(b0 + 2), P(b0 + 3), midb, MTILES);
    gemm_node_kernel<1><<<nodeBlocks, 128, 0, stream>>>(
        midb, Wn2[l], P(b0 + 5), nullptr, nullptr, hnb, MTILES);
    zero_f32_kernel<<<(NN * 128 + 255) / 256, 256, 0, stream>>>(hagg, NN * 128);
    edge_kernel<<<edgeBlocks, 128, 0, stream>>>(
        hbuf, srcI, dstI, eattr, We1[l], P(b0 + 7), P(b0 + 8), P(b0 + 9), We2[l],
        P(b0 + 11), hagg);
    update_kernel<<<nodeBlocks, 128, 0, stream>>>(
        hnb, hagg, Wup[l], P(b0 + 13), P(b0 + 14), P(b0 + 15), h, hbuf, MTILES);
  }

  // --- output head + pooling ---
  zero_f32_kernel<<<(16 * 128 + 255) / 256, 256, 0, stream>>>(pool_add, 16 * 128);
  zero_f32_kernel<<<1, 64, 0, stream>>>(pool_cnt, 16);
  out_kernel<<<nodeBlocks, 128, 0, stream>>>(hbuf, Wout, P(58), P(59), P(60), batch,
                                             (float*)d_out, pool_add, pool_cnt,
                                             MTILES);
  pool_kernel<<<1, 128, 0, stream>>>(pool_add, pool_cnt, P(61), P(62),
                                     (float*)d_out + (size_t)NN * 128);
}
